// CGCLayer_11235634446410
// MI455X (gfx1250) — matrix operand order repaired and checked by reading, not compiled, not run on hardware
//
#include <hip/hip_runtime.h>
#include <math.h>

// ---------------------------------------------------------------------------
// CGC layer for MI455X (gfx1250): f16 WMMA GEMM (f32 accumulate), 160 KB LDS
// weight-fragment cache, M-blocked (2 edge tiles / wave) so each B fragment
// read from LDS feeds 2 WMMAs, f32 atomic scatter-sum.
// ---------------------------------------------------------------------------

typedef _Float16 v16h __attribute__((ext_vector_type(16)));
typedef _Float16 v8h  __attribute__((ext_vector_type(8)));
typedef float    v8f  __attribute__((ext_vector_type(8)));

#define NODE_DIM 128
#define EDGE_DIM 64
#define HID      320
#define KT       10          // K tiles of 32 (320/32)
#define NT       8           // N tiles of 16 (128/16)
#define NFRAG    (2*KT*NT)   // 160 B-fragments (val+mul)
#define FRAG_F16 512         // 32x16 f16 per fragment
#define BS_ELEMS (NFRAG*FRAG_F16)  // 81920 f16 = 160 KB LDS

__device__ __forceinline__ float softplus_f(float x) {
    return fmaxf(x, 0.0f) + log1pf(__expf(-fabsf(x)));   // stable log1p(exp(x))
}
__device__ __forceinline__ float sigmoid_f(float x) {
    return __builtin_amdgcn_rcpf(1.0f + __expf(-x));
}

__global__ __launch_bounds__(256)
void cgc_wmma_kernel(const float* __restrict__ x,
                     const int*   __restrict__ ei,      // (2, E)
                     const float* __restrict__ ef,      // (E, 64)
                     const float* __restrict__ Wv,      // (128, 320)
                     const float* __restrict__ bv,      // (128)
                     const float* __restrict__ Wm,      // (128, 320)
                     const float* __restrict__ bm,      // (128)
                     float* __restrict__ out,           // (N, 128)
                     int E, int nPairs)
{
    // 160 KB of LDS: legal on CDNA5 (320 KB/WGP).
    __shared__ __align__(16) _Float16 Bs[BS_ELEMS];

    const int tid = threadIdx.x;

    // ---- Stage both weight matrices as WMMA B fragments (f32 -> f16) ------
    // Fragment id f = nt*20 + kt*2 + w  (w: 0=val, 1=mul)
    // B is KxN = 32x16 with B[k][n] = W[nt*16+n][kt*32+k].
    // Lane l (g = l>>4): n = l&15; elems 0..7 -> k = 8g+e, elems 8..15 -> k = 16+8g+(e-8).
    // Linear layout: Bs[f*512 + hh*256 + lane*8 + e8].
    for (int c = tid; c < BS_ELEMS / 8; c += blockDim.x) {
        const int l  = c & 31;
        const int hh = (c >> 5) & 1;
        const int f  = c >> 6;
        const int w  = f & 1;
        const int kt = (f >> 1) % KT;
        const int nt = f / (2 * KT);
        const int n  = nt * 16 + (l & 15);
        const int kb = kt * 32 + (l >> 4) * 8 + hh * 16;
        const float* W = w ? Wm : Wv;
        const float* src = W + (size_t)n * HID + kb;
        float4 s0 = *(const float4*)(src);
        float4 s1 = *(const float4*)(src + 4);
        v8h p;
        p[0] = (_Float16)s0.x; p[1] = (_Float16)s0.y;
        p[2] = (_Float16)s0.z; p[3] = (_Float16)s0.w;
        p[4] = (_Float16)s1.x; p[5] = (_Float16)s1.y;
        p[6] = (_Float16)s1.z; p[7] = (_Float16)s1.w;
        *(v8h*)(&Bs[c * 8]) = p;
    }
    __syncthreads();

    const int lane = tid & 31;
    const int wave = tid >> 5;
    const int g    = lane >> 4;   // lane group
    const int m    = lane & 15;   // my A-row within a tile
    const int wavesInGrid = gridDim.x * (blockDim.x >> 5);

    for (int p = blockIdx.x * (blockDim.x >> 5) + wave; p < nPairs; p += wavesInGrid) {
        const int base0 = p * 32;         // first edge of tile 0
        const int base1 = base0 + 16;     // first edge of tile 1
        const int e0  = base0 + m;
        const int e1  = base1 + m;
        const int el0 = (e0 < E) ? e0 : (E - 1);
        const int el1 = (e1 < E) ? e1 : (E - 1);
        const int s0i = ei[el0],     s1i = ei[el1];
        const int r0i = ei[E + el0], r1i = ei[E + el1];

        // ---- Build 2 x 10 A fragments (16x32 f16) in registers ------------
        // A layout (16-bit, 16x32): lane m; elems 0..7 -> k = 8g..8g+7,
        //                           elems 8..15 -> k = 16+8g..23+8g.
        const int c0 = 8 * g;
        v16h a0[KT], a1[KT];
        {
            const float* xs0 = x  + (size_t)s0i * NODE_DIM;
            const float* xr0 = x  + (size_t)r0i * NODE_DIM;
            const float* ep0 = ef + (size_t)el0 * EDGE_DIM;
            const float* xs1 = x  + (size_t)s1i * NODE_DIM;
            const float* xr1 = x  + (size_t)r1i * NODE_DIM;
            const float* ep1 = ef + (size_t)el1 * EDGE_DIM;
            #pragma unroll
            for (int kt = 0; kt < KT; ++kt) {
                const float* b0 = (kt < 4) ? (xs0 + kt * 32)
                                : (kt < 8) ? (xr0 + (kt - 4) * 32)
                                           : (ep0 + (kt - 8) * 32);
                const float* b1 = (kt < 4) ? (xs1 + kt * 32)
                                : (kt < 8) ? (xr1 + (kt - 4) * 32)
                                           : (ep1 + (kt - 8) * 32);
                float fa[16], fb[16];
                *(float4*)(&fa[0])  = *(const float4*)(b0 + c0);
                *(float4*)(&fa[4])  = *(const float4*)(b0 + c0 + 4);
                *(float4*)(&fa[8])  = *(const float4*)(b0 + 16 + c0);
                *(float4*)(&fa[12]) = *(const float4*)(b0 + 16 + c0 + 4);
                *(float4*)(&fb[0])  = *(const float4*)(b1 + c0);
                *(float4*)(&fb[4])  = *(const float4*)(b1 + c0 + 4);
                *(float4*)(&fb[8])  = *(const float4*)(b1 + 16 + c0);
                *(float4*)(&fb[12]) = *(const float4*)(b1 + 16 + c0 + 4);
                v16h av, bvv;
                #pragma unroll
                for (int q = 0; q < 16; ++q) { av[q] = (_Float16)fa[q]; bvv[q] = (_Float16)fb[q]; }
                a0[kt] = av;
                a1[kt] = bvv;
            }
        }

        // ---- Receiver row ids for my 8 C-rows per tile (M = i + 8g) -------
        int rr[16];
        #pragma unroll
        for (int i = 0; i < 8; ++i) {
            rr[i]     = __shfl(r0i, i + 8 * g, 32);
            rr[8 + i] = __shfl(r1i, i + 8 * g, 32);
        }

        const bool full = (base0 + 32 <= E);   // wave-uniform fast path

        // ---- N-tile loop: each B fragment pair feeds 4 WMMAs --------------
        for (int nt = 0; nt < NT; ++nt) {
            v8f av0 = {0.f,0.f,0.f,0.f,0.f,0.f,0.f,0.f};
            v8f am0 = {0.f,0.f,0.f,0.f,0.f,0.f,0.f,0.f};
            v8f av1 = {0.f,0.f,0.f,0.f,0.f,0.f,0.f,0.f};
            v8f am1 = {0.f,0.f,0.f,0.f,0.f,0.f,0.f,0.f};
            #pragma unroll
            for (int kt = 0; kt < KT; ++kt) {
                const int f = nt * 20 + kt * 2;
                const _Float16* bp = &Bs[f * FRAG_F16 + lane * 8];
                v8h vl = *(const v8h*)(bp);
                v8h vh = *(const v8h*)(bp + 256);
                v16h bfv = __builtin_shufflevector(vl, vh,
                    0,1,2,3,4,5,6,7,8,9,10,11,12,13,14,15);
                v8h ml = *(const v8h*)(bp + FRAG_F16);
                v8h mh = *(const v8h*)(bp + FRAG_F16 + 256);
                v16h bfm = __builtin_shufflevector(ml, mh,
                    0,1,2,3,4,5,6,7,8,9,10,11,12,13,14,15);
                av0 = __builtin_amdgcn_wmma_f32_16x16x32_f16(
                    false, a0[kt], false, bfv, (short)0, av0, false, false);
                av1 = __builtin_amdgcn_wmma_f32_16x16x32_f16(
                    false, a1[kt], false, bfv, (short)0, av1, false, false);
                am0 = __builtin_amdgcn_wmma_f32_16x16x32_f16(
                    false, a0[kt], false, bfm, (short)0, am0, false, false);
                am1 = __builtin_amdgcn_wmma_f32_16x16x32_f16(
                    false, a1[kt], false, bfm, (short)0, am1, false, false);
            }
            // C/D layout: lane column n = nt*16 + m; VGPR i -> row M = i + 8g.
            const int col = nt * 16 + m;
            const float bvs = bv[col];
            const float bms = bm[col];
            if (full) {
                #pragma unroll
                for (int i = 0; i < 8; ++i) {
                    const float v0  = softplus_f(av0[i] + bvs) * sigmoid_f(am0[i] + bms);
                    const float v1  = softplus_f(av1[i] + bvs) * sigmoid_f(am1[i] + bms);
                    unsafeAtomicAdd(out + (size_t)rr[i]     * NODE_DIM + col, v0);
                    unsafeAtomicAdd(out + (size_t)rr[8 + i] * NODE_DIM + col, v1);
                }
            } else {
                #pragma unroll
                for (int i = 0; i < 8; ++i) {
                    if (base0 + i + 8 * g < E) {
                        const float v0 = softplus_f(av0[i] + bvs) * sigmoid_f(am0[i] + bms);
                        unsafeAtomicAdd(out + (size_t)rr[i] * NODE_DIM + col, v0);
                    }
                    if (base1 + i + 8 * g < E) {
                        const float v1 = softplus_f(av1[i] + bvs) * sigmoid_f(am1[i] + bms);
                        unsafeAtomicAdd(out + (size_t)rr[8 + i] * NODE_DIM + col, v1);
                    }
                }
            }
        }
    }
}

extern "C" void kernel_launch(void* const* d_in, const int* in_sizes, int n_in,
                              void* d_out, int out_size, void* d_ws, size_t ws_size,
                              hipStream_t stream) {
    const float* x  = (const float*)d_in[0];
    const int*   ei = (const int*)  d_in[1];
    const float* ef = (const float*)d_in[2];
    const float* Wv = (const float*)d_in[3];
    const float* bv = (const float*)d_in[4];
    const float* Wm = (const float*)d_in[5];
    const float* bm = (const float*)d_in[6];
    float* out = (float*)d_out;

    const int E      = in_sizes[1] / 2;
    const int nTiles = (E + 15) / 16;
    const int nPairs = (nTiles + 1) / 2;

    // Output is scatter-accumulated: zero it (capture-safe async memset).
    hipMemsetAsync(out, 0, (size_t)out_size * sizeof(float), stream);

    dim3 block(256);                 // 8 wave32 waves per block
    dim3 grid(640);                  // grid-stride over 25000 tile-pairs
    hipLaunchKernelGGL(cgc_wmma_kernel, grid, block, 0, stream,
                       x, ei, ef, Wv, bv, Wm, bm, out, E, nPairs);
}
